// Wavenet_24962349924850
// MI455X (gfx1250) — compile-verified, matrix operand order checked by reading, not run on hardware
//
#include <hip/hip_runtime.h>

// ---------------------------------------------------------------------------
// WaveNet forward on gfx1250 (MI455X), bf16 WMMA everywhere, per-layer fused.
//
// Layouts:
//   activations  h             :  [col][chan]   (col = n*16384 + t), bf16
//   skip accumulator           :  [col][256]    f32
//   weights (prepped)          :  row-major [M][K], bf16
//
// WMMA v_wmma_f32_16x16x32_bf16 fragment layout (wave32):
//   A (16x32, MxK): lane L holds row M=L%16; K-half = (L<16?0:8);
//                   vector elem j (j<8): K = Khalf + j ; (j>=8): K = 16+Khalf+(j-8)
//   => one fragment = two contiguous 16B loads from a row-major [M][K] matrix.
//   B (32x16, KxN): lane L holds column N=L%16, same K packing
//   => one fragment = two contiguous 16B loads from a col-major-[col][K] buffer.
//   C/D (16x16 f32): VGPR v, lanes 0-15: (M=v, N=lane); lanes 16-31: (M=v+8).
//
// Per-layer kernel is fused: conv(2-tap,dilated)+GLU -> 8KB LDS tile ->
// barrier -> res/skip GEMM with B-fragments from ds_load_b128.
//
// Workspace requirement: ~104 MB (hA,hB 16MB each; skip 64MB; weights ~8MB).
// ---------------------------------------------------------------------------

typedef __attribute__((ext_vector_type(16))) __bf16 v16bf;
typedef __attribute__((ext_vector_type(8)))  float  v8f;

#define NCOL 65536   // 4 batches * 16384 samples
#define WLEN 16384
#define CRES 128
#define C2   256
#define NLAY 30

union Frag16 { v16bf v; uint4 q[2]; __bf16 e[16]; };

__device__ __forceinline__ v16bf frag_ld(const __bf16* p) {
  Frag16 f;
  f.q[0] = *reinterpret_cast<const uint4*>(p);
  f.q[1] = *reinterpret_cast<const uint4*>(p + 16);
  return f.v;
}

__device__ __forceinline__ v16bf frag_ld_guard(const __bf16* p, bool valid) {
  Frag16 f;
  if (valid) {
    f.q[0] = *reinterpret_cast<const uint4*>(p);
    f.q[1] = *reinterpret_cast<const uint4*>(p + 16);
  } else {
    f.q[0] = make_uint4(0u, 0u, 0u, 0u);
    f.q[1] = make_uint4(0u, 0u, 0u, 0u);
  }
  return f.v;
}

// load 16 K-values from an f32 buffer, relu, convert to bf16 fragment
__device__ __forceinline__ v16bf frag_ld_f32_relu(const float* p) {
  Frag16 f;
#pragma unroll
  for (int j = 0; j < 8; ++j) f.e[j]     = (__bf16)fmaxf(p[j], 0.f);
#pragma unroll
  for (int j = 0; j < 8; ++j) f.e[8 + j] = (__bf16)fmaxf(p[16 + j], 0.f);
  return f.v;
}

__device__ __forceinline__ v8f wmma_bf16(v16bf a, v16bf b, v8f c) {
  return __builtin_amdgcn_wmma_f32_16x16x32_bf16(false, a, false, b,
                                                 (short)0, c, false, false);
}

__device__ __forceinline__ v8f vzero8() {
  v8f z = {0.f, 0.f, 0.f, 0.f, 0.f, 0.f, 0.f, 0.f};
  return z;
}

// ---------------------------------------------------------------------------
// Front: h0[col][c] = b_shift[c] + w0[c]*x[t-2] + w1[c]*x[t-1]
// (input shifted right by one, then causal k=2 conv)
// ---------------------------------------------------------------------------
__global__ __launch_bounds__(256) void k_front(const float* __restrict__ x,
                                               const float* __restrict__ w_shift,
                                               const float* __restrict__ b_shift,
                                               __bf16* __restrict__ h) {
  int col = blockIdx.x * 256 + threadIdx.x;
  if (col >= NCOL) return;
  int t = col & (WLEN - 1);
  int n = col >> 14;
  const float* xb = x + (size_t)n * WLEN;
  float x0 = (t >= 2) ? xb[t - 2] : 0.f;
  float x1 = (t >= 1) ? xb[t - 1] : 0.f;
  __bf16* hp = h + (size_t)col * CRES;
  for (int c0 = 0; c0 < CRES; c0 += 8) {
    union { uint4 q; __bf16 b[8]; } o;
#pragma unroll
    for (int j = 0; j < 8; ++j) {
      int c = c0 + j;
      o.b[j] = (__bf16)(w_shift[c * 2 + 0] * x0 + w_shift[c * 2 + 1] * x1 + b_shift[c]);
    }
    *reinterpret_cast<uint4*>(hp + c0) = o.q;
  }
}

// ---------------------------------------------------------------------------
// Weight prep: fold the two conv taps into one 256x256 matrix per layer.
// wcat[l][o][k] : k<128 -> tap0 (applies to col-d), k>=128 -> tap1 (col)
// ---------------------------------------------------------------------------
__global__ void k_prep_wcat(const float* __restrict__ conv_w, __bf16* __restrict__ wcat,
                            int total) {
  int i = blockIdx.x * 256 + threadIdx.x;
  if (i >= total) return;
  int k = i & 255;
  int o = (i >> 8) & 255;
  int l = i >> 16;
  int c = k & 127;
  int tap = k >> 7;
  wcat[i] = (__bf16)conv_w[(((size_t)l * C2 + o) * CRES + c) * 2 + tap];
}

// wrs[l][m][r]: m<128 -> res_w[l][m][r], m>=128 -> skip_w[l][m-128][r]
__global__ void k_prep_wrs(const float* __restrict__ res_w, const float* __restrict__ skip_w,
                           __bf16* __restrict__ wrs, int total) {
  int i = blockIdx.x * 256 + threadIdx.x;
  if (i >= total) return;
  int r = i & 127;
  int m = (i >> 7) % 384;
  int l = i / (384 * 128);
  float v;
  if (m < 128) v = res_w[((size_t)l * 128 + m) * 128 + r];
  else         v = skip_w[((size_t)l * 256 + (m - 128)) * 128 + r];
  wrs[i] = (__bf16)v;
}

__global__ void k_cvt_f32_bf16(const float* __restrict__ s, __bf16* __restrict__ d, int n) {
  int i = blockIdx.x * 256 + threadIdx.x;
  if (i < n) d[i] = (__bf16)s[i];
}

// ---------------------------------------------------------------------------
// Fused residual layer.  Workgroup = 8 waves, N = 32 cols.
//
// Phase 1 (conv+GLU): G = Wcat(256x256) @ Xcat(256 x 32) + conv_b,
//   Xcat rows 0..127 = h[col-d] (zero if t<d), rows 128..255 = h[col].
//   Wave w owns row-tile pair (w, w+8) so it holds both GLU halves:
//   gated[col][r] = G[r]*sigmoid(G[r+128])  ->  8KB LDS tile [32 cols][128].
//
// Phase 2 (after barrier): R = [Wres;Wskip](384x128) @ gated_LDS(128x32)
//   wave w owns row-tiles {w, w+8, w+16} x 2 col-tiles; B-frags via ds_load.
//   hOut[col][c]  = bf16(R[c] + res_b[c] + hIn[col][c])       (c < 128)
//   skip[col][s] += R[128+s] + skip_b[s]   (plain write if firstLayer)
// ---------------------------------------------------------------------------
__global__ __launch_bounds__(256) void k_layer(const __bf16* __restrict__ hIn,
                                               const __bf16* __restrict__ wcat,
                                               const __bf16* __restrict__ wrs,
                                               const float* __restrict__ cb,
                                               const float* __restrict__ res_b,
                                               const float* __restrict__ skip_b,
                                               __bf16* __restrict__ hOut,
                                               float* __restrict__ skip,
                                               int dil, int firstLayer) {
  __shared__ __align__(16) __bf16 lg[32 * CRES];   // gated tile [col_local][chan]

  const int wave = threadIdx.x >> 5;
  const int lane = threadIdx.x & 31;
  const int l15  = lane & 15;
  const int kh   = (lane >> 4) << 3;               // K half: 0 or 8
  const int mb   = (lane >> 4) * 8;                // D-row offset within tile
  const int colBase = blockIdx.x * 32;

  // ---------------- phase 1: dilated conv + GLU ----------------
  {
    const int rtA = wave;                          // rows rtA*16..  (GLU 'a')
    const int rtB = wave + 8;                      // rows +128      (GLU gate)
    const __bf16* wA = wcat + ((size_t)(rtA * 16 + l15)) * C2;
    const __bf16* wB = wcat + ((size_t)(rtB * 16 + l15)) * C2;

    int colT0 = colBase + l15;
    int colT1 = colBase + 16 + l15;
    bool val0 = (colT0 & (WLEN - 1)) >= dil;       // 32-col tile never crosses batch
    bool val1 = (colT1 & (WLEN - 1)) >= dil;

    v8f accA0 = vzero8(), accA1 = vzero8(), accB0 = vzero8(), accB1 = vzero8();

#pragma unroll
    for (int ks = 0; ks < 8; ++ks) {
      const int k0 = ks * 32;
      v16bf a0 = frag_ld(wA + k0 + kh);
      v16bf a1 = frag_ld(wB + k0 + kh);
      v16bf b0, b1;
      if (ks < 4) {                                // tap0: column col-d, guarded
        b0 = frag_ld_guard(hIn + (size_t)(colT0 - dil) * CRES + k0 + kh, val0);
        b1 = frag_ld_guard(hIn + (size_t)(colT1 - dil) * CRES + k0 + kh, val1);
      } else {                                     // tap1: column col
        b0 = frag_ld(hIn + (size_t)colT0 * CRES + (k0 - 128) + kh);
        b1 = frag_ld(hIn + (size_t)colT1 * CRES + (k0 - 128) + kh);
      }
      accA0 = wmma_bf16(a0, b0, accA0);
      accA1 = wmma_bf16(a0, b1, accA1);
      accB0 = wmma_bf16(a1, b0, accB0);
      accB1 = wmma_bf16(a1, b1, accB1);
    }

    const int rowA = rtA * 16 + mb;                // channel base 0..120
#pragma unroll
    for (int ct = 0; ct < 2; ++ct) {
      int cl = ct * 16 + l15;                      // col_local 0..31
      union { uint4 q; __bf16 b[8]; } o;
#pragma unroll
      for (int v = 0; v < 8; ++v) {
        float ga = (ct ? accA1[v] : accA0[v]) + cb[rowA + v];
        float gb = (ct ? accB1[v] : accB0[v]) + cb[rowA + 128 + v];
        o.b[v] = (__bf16)(ga / (1.f + __expf(-gb)));   // a * sigmoid(b)
      }
      *reinterpret_cast<uint4*>(&lg[cl * CRES + rowA]) = o.q;
    }
  }

  __syncthreads();

  // ---------------- phase 2: res + skip GEMM from LDS ----------------
  {
    v8f acc[3][2];
    const __bf16* wp[3];
#pragma unroll
    for (int i = 0; i < 3; ++i) {
      acc[i][0] = vzero8();
      acc[i][1] = vzero8();
      wp[i] = wrs + ((size_t)((wave + i * 8) * 16 + l15)) * CRES;
    }

#pragma unroll
    for (int ks = 0; ks < 4; ++ks) {
      const int k0 = ks * 32;
      v16bf b0 = frag_ld(&lg[(l15) * CRES + k0 + kh]);        // ds_load_b128 x2
      v16bf b1 = frag_ld(&lg[(16 + l15) * CRES + k0 + kh]);
#pragma unroll
      for (int i = 0; i < 3; ++i) {
        v16bf a = frag_ld(wp[i] + k0 + kh);
        acc[i][0] = wmma_bf16(a, b0, acc[i][0]);
        acc[i][1] = wmma_bf16(a, b1, acc[i][1]);
      }
    }

#pragma unroll
    for (int i = 0; i < 3; ++i) {
      int m0 = (wave + i * 8) * 16 + mb;
#pragma unroll
      for (int ct = 0; ct < 2; ++ct) {
        int col = colBase + ct * 16 + l15;
        if (m0 < 128) {                            // residual path -> next h
          union { uint4 q; __bf16 b[8]; } hin, o;
          hin.q = *reinterpret_cast<const uint4*>(hIn + (size_t)col * CRES + m0);
#pragma unroll
          for (int v = 0; v < 8; ++v)
            o.b[v] = (__bf16)(acc[i][ct][v] + res_b[m0 + v] + (float)hin.b[v]);
          *reinterpret_cast<uint4*>(hOut + (size_t)col * CRES + m0) = o.q;
        } else {                                   // skip accumulation (f32)
          int s0 = m0 - 128;
          float* sp = skip + (size_t)col * C2 + s0;
          if (firstLayer) {
#pragma unroll
            for (int v = 0; v < 8; ++v) sp[v] = acc[i][ct][v] + skip_b[s0 + v];
          } else {
#pragma unroll
            for (int v = 0; v < 8; ++v) sp[v] += acc[i][ct][v] + skip_b[s0 + v];
          }
        }
      }
    }
  }
}

// ---------------------------------------------------------------------------
// Fused head.  Workgroup = 8 waves, N = 16 cols, K = 256 both GEMMs.
// Phase 1: z = relu(a_w @ relu(skip) + a_b) -> 8KB LDS tile [16 cols][256].
// Phase 2: logits = b_w @ z + b_b, scattered to (n, 256, 1, 16384) f32.
// ---------------------------------------------------------------------------
__global__ __launch_bounds__(256) void k_head(const float* __restrict__ skip,
                                              const __bf16* __restrict__ aw,
                                              const float* __restrict__ ab,
                                              const __bf16* __restrict__ bw,
                                              const float* __restrict__ bb,
                                              float* __restrict__ out) {
  __shared__ __align__(16) __bf16 lz[16 * C2];     // z tile [col_local][256]

  const int wave = threadIdx.x >> 5;
  const int lane = threadIdx.x & 31;
  const int l15  = lane & 15;
  const int kh   = (lane >> 4) << 3;
  const int mb   = (lane >> 4) * 8;
  const int col  = blockIdx.x * 16 + l15;

  // ---------------- phase 1: z = relu(a_w @ relu(skip) + a_b) ----------------
  {
    const float* bp = skip + (size_t)col * C2 + kh;
    const __bf16* w0 = aw + ((size_t)(wave * 16 + l15)) * C2;
    const __bf16* w1 = aw + ((size_t)((wave + 8) * 16 + l15)) * C2;

    v8f acc0 = vzero8(), acc1 = vzero8();
#pragma unroll
    for (int ks = 0; ks < 8; ++ks) {
      const int k0 = ks * 32;
      v16bf b = frag_ld_f32_relu(bp + k0);
      acc0 = wmma_bf16(frag_ld(w0 + k0 + kh), b, acc0);
      acc1 = wmma_bf16(frag_ld(w1 + k0 + kh), b, acc1);
    }

#pragma unroll
    for (int i = 0; i < 2; ++i) {
      int m0 = (wave + i * 8) * 16 + mb;
      union { uint4 q; __bf16 b[8]; } o;
#pragma unroll
      for (int v = 0; v < 8; ++v)
        o.b[v] = (__bf16)fmaxf((i ? acc1[v] : acc0[v]) + ab[m0 + v], 0.f);
      *reinterpret_cast<uint4*>(&lz[l15 * C2 + m0]) = o.q;
    }
  }

  __syncthreads();

  // ---------------- phase 2: logits = b_w @ z + b_b ----------------
  {
    const __bf16* bp = &lz[l15 * C2 + kh];
    const __bf16* w0 = bw + ((size_t)(wave * 16 + l15)) * C2;
    const __bf16* w1 = bw + ((size_t)((wave + 8) * 16 + l15)) * C2;

    v8f acc0 = vzero8(), acc1 = vzero8();
#pragma unroll
    for (int ks = 0; ks < 8; ++ks) {
      const int k0 = ks * 32;
      v16bf b = frag_ld(bp + k0);                  // ds_load_b128 x2
      acc0 = wmma_bf16(frag_ld(w0 + k0 + kh), b, acc0);
      acc1 = wmma_bf16(frag_ld(w1 + k0 + kh), b, acc1);
    }

    const int n = col >> 14;
    const int t = col & (WLEN - 1);
#pragma unroll
    for (int i = 0; i < 2; ++i) {
      int m0 = (wave + i * 8) * 16 + mb;
#pragma unroll
      for (int v = 0; v < 8; ++v) {
        int o = m0 + v;
        out[((size_t)n * C2 + o) * WLEN + t] = (i ? acc1[v] : acc0[v]) + bb[o];
      }
    }
  }
}

// ---------------------------------------------------------------------------
extern "C" void kernel_launch(void* const* d_in, const int* in_sizes, int n_in,
                              void* d_out, int out_size, void* d_ws, size_t ws_size,
                              hipStream_t stream) {
  const float* x       = (const float*)d_in[0];
  const float* w_shift = (const float*)d_in[1];
  const float* b_shift = (const float*)d_in[2];
  const float* conv_w  = (const float*)d_in[3];
  const float* conv_b  = (const float*)d_in[4];
  const float* res_w   = (const float*)d_in[5];
  const float* res_b   = (const float*)d_in[6];
  const float* skip_w  = (const float*)d_in[7];
  const float* skip_b  = (const float*)d_in[8];
  const float* a_w     = (const float*)d_in[9];
  const float* a_b     = (const float*)d_in[10];
  const float* b_w     = (const float*)d_in[11];
  const float* b_b     = (const float*)d_in[12];
  (void)in_sizes; (void)n_in; (void)out_size; (void)ws_size;

  char* ws = (char*)d_ws;
  size_t off = 0;
  __bf16* hA   = (__bf16*)(ws + off); off += (size_t)NCOL * CRES * 2;  // 16 MB
  __bf16* hB   = (__bf16*)(ws + off); off += (size_t)NCOL * CRES * 2;  // 16 MB
  float*  skip = (float*) (ws + off); off += (size_t)NCOL * C2 * 4;    // 64 MB
  __bf16* wcat = (__bf16*)(ws + off); off += (size_t)NLAY * C2 * C2 * 2;
  __bf16* wrs  = (__bf16*)(ws + off); off += (size_t)NLAY * 384 * CRES * 2;
  __bf16* awb  = (__bf16*)(ws + off); off += (size_t)C2 * C2 * 2;
  __bf16* bwb  = (__bf16*)(ws + off); off += (size_t)C2 * C2 * 2;

  // ---- prep ----
  k_front<<<NCOL / 256, 256, 0, stream>>>(x, w_shift, b_shift, hA);
  {
    int tot = NLAY * C2 * C2;
    k_prep_wcat<<<(tot + 255) / 256, 256, 0, stream>>>(conv_w, wcat, tot);
  }
  {
    int tot = NLAY * 384 * CRES;
    k_prep_wrs<<<(tot + 255) / 256, 256, 0, stream>>>(res_w, skip_w, wrs, tot);
  }
  k_cvt_f32_bf16<<<(C2 * C2 + 255) / 256, 256, 0, stream>>>(a_w, awb, C2 * C2);
  k_cvt_f32_bf16<<<(C2 * C2 + 255) / 256, 256, 0, stream>>>(b_w, bwb, C2 * C2);

  // ---- residual stack (fused per-layer kernel) ----
  __bf16* hin = hA;
  __bf16* hout = hB;
  for (int l = 0; l < NLAY; ++l) {
    int dil = 1 << (l % 10);
    k_layer<<<NCOL / 32, 256, 0, stream>>>(hin,
                                           wcat + (size_t)l * C2 * C2,
                                           wrs + (size_t)l * 384 * CRES,
                                           conv_b + (size_t)l * C2,
                                           res_b + (size_t)l * CRES,
                                           skip_b + (size_t)l * C2,
                                           hout, skip, dil, l == 0 ? 1 : 0);
    __bf16* tmp = hin; hin = hout; hout = tmp;
  }

  // ---- fused head ----
  k_head<<<NCOL / 16, 256, 0, stream>>>(skip, awb, a_b, bwb, b_b, (float*)d_out);
}
